// HDRNetwoBN_49898930045559
// MI455X (gfx1250) — matrix operand order repaired and checked by reading, not compiled, not run on hardware
//
#include <hip/hip_runtime.h>
#include <hip/hip_bf16.h>
#include <stdint.h>

typedef __attribute__((ext_vector_type(16))) _Float16 v16h;
typedef __attribute__((ext_vector_type(8)))  float    v8f;
typedef __attribute__((__vector_size__(4 * sizeof(int)))) int v4i;

#define GD 8
#define GH 16
#define GW 16
#define FULL_H 1024
#define FULL_W 1024
#define CG_FLOATS (GD * GH * GW * 6)     // 12288 floats = 48 KB
#define TILE_ROWS 34
#define TILE_W    36                     // padded stride

#if __has_builtin(__builtin_amdgcn_global_load_async_to_lds_b128) && \
    __has_builtin(__builtin_amdgcn_s_wait_asynccnt)
#define USE_ASYNC_LDS 1
#endif

// ---------------------------------------------------------------------------
// Generic direct 3x3 conv, pad=1, stride 1 or 2, optional bias / relu.
// One thread per output element. Only used for the tiny low-res tensors.
// ---------------------------------------------------------------------------
__global__ void conv3x3_kernel(const float* __restrict__ in,
                               const float* __restrict__ w,
                               const float* __restrict__ b,
                               float* __restrict__ out,
                               int N, int Ci, int Hi, int Wi,
                               int Co, int Ho, int Wo,
                               int stride, int do_relu)
{
    int idx = blockIdx.x * blockDim.x + threadIdx.x;
    int total = N * Co * Ho * Wo;
    if (idx >= total) return;
    int wo = idx % Wo;
    int ho = (idx / Wo) % Ho;
    int co = (idx / (Wo * Ho)) % Co;
    int n  = idx / (Wo * Ho * Co);

    float acc = b ? b[co] : 0.0f;
    for (int ci = 0; ci < Ci; ++ci) {
        const float* ip = in + ((long)(n * Ci + ci)) * Hi * Wi;
        const float* wp = w + ((co * Ci + ci) * 9);
        for (int ky = 0; ky < 3; ++ky) {
            int yy = ho * stride + ky - 1;
            if (yy < 0 || yy >= Hi) continue;
            for (int kx = 0; kx < 3; ++kx) {
                int xx = wo * stride + kx - 1;
                if (xx < 0 || xx >= Wi) continue;
                acc += ip[yy * Wi + xx] * wp[ky * 3 + kx];
            }
        }
    }
    if (do_relu) acc = fmaxf(acc, 0.0f);
    out[idx] = acc;
}

// ---------------------------------------------------------------------------
// Global branch tail: pooled means -> FC 448->256->128->64 -> fuse with local
// -> 1x1 conv to 96ch grid -> pre-combined 6-channel grid.
// One block per batch image, 256 threads.
// cgrid layout: [n][d][gy][gx][6]
// ---------------------------------------------------------------------------
__global__ void __launch_bounds__(256)
global_tail_kernel(const float* __restrict__ splat,  // (N,64,16,16)
                   const float* __restrict__ gx1,    // (N,128,8,8)
                   const float* __restrict__ gx2,    // (N,256,4,4)
                   const float* __restrict__ local,  // (N,64,16,16)
                   const float* __restrict__ wf1, const float* __restrict__ bf1,
                   const float* __restrict__ wf2, const float* __restrict__ bf2,
                   const float* __restrict__ wf3, const float* __restrict__ bf3,
                   const float* __restrict__ wlin, const float* __restrict__ blin,
                   float* __restrict__ cgrid)        // (N,8,16,16,6)
{
    __shared__ float a[448];
    __shared__ float h1[256];
    __shared__ float h2[128];
    __shared__ float glob[64];

    const int n = blockIdx.x;
    const int tid = threadIdx.x;

    if (tid < 64) {
        float s = 0.0f;
        const float* p = splat + (n * 64 + tid) * 256;
        for (int i = 0; i < 256; ++i) s += p[i];
        a[tid] = s * (1.0f / 256.0f);
    }
    if (tid < 128) {
        float s = 0.0f;
        const float* p = gx1 + (n * 128 + tid) * 64;
        for (int i = 0; i < 64; ++i) s += p[i];
        a[64 + tid] = s * (1.0f / 64.0f);
    }
    {
        float s = 0.0f;
        const float* p = gx2 + (n * 256 + tid) * 16;
        for (int i = 0; i < 16; ++i) s += p[i];
        a[192 + tid] = s * (1.0f / 16.0f);
    }
    __syncthreads();
    {
        float s = bf1[tid];
        const float* wr = wf1 + tid * 448;
        for (int i = 0; i < 448; ++i) s += wr[i] * a[i];
        h1[tid] = fmaxf(s, 0.0f);
    }
    __syncthreads();
    if (tid < 128) {
        float s = bf2[tid];
        const float* wr = wf2 + tid * 256;
        for (int i = 0; i < 256; ++i) s += wr[i] * h1[i];
        h2[tid] = fmaxf(s, 0.0f);
    }
    __syncthreads();
    if (tid < 64) {
        float s = bf3[tid];
        const float* wr = wf3 + tid * 128;
        for (int i = 0; i < 128; ++i) s += wr[i] * h2[i];
        glob[tid] = fmaxf(s, 0.0f);
    }
    __syncthreads();

    // Each thread owns one of the 16x16 grid cells.
    const int y = tid >> 4;
    const int x = tid & 15;
    float fused[64];
    for (int c = 0; c < 64; ++c)
        fused[c] = fmaxf(glob[c] + local[((n * 64 + c) * 16 + y) * 16 + x], 0.0f);

    for (int d = 0; d < 8; ++d) {
        float v[12];
        for (int c12 = 0; c12 < 12; ++c12) {
            int o = c12 * 8 + d;                 // reshape (96) -> (12,8)
            float s = blin[o];
            const float* wr = wlin + o * 64;
            for (int c = 0; c < 64; ++c) s += wr[c] * fused[c];
            v[c12] = s;
        }
        float* dst = cgrid + ((((n * 8 + d) * 16 + y) * 16 + x) * 6);
        dst[0] = v[0] + v[1] + v[2];
        dst[1] = v[3];
        dst[2] = v[4] + v[5] + v[6];
        dst[3] = v[7];
        dst[4] = v[8] + v[9] + v[10];
        dst[5] = v[11];
    }
}

// ---------------------------------------------------------------------------
// Fused full-res kernel:
//   guide = tanh(1x1(relu(3x3(full))))  via v_wmma_f32_16x16x32_f16
//   coeff = trilinear slice of pre-combined grid (LDS resident)
//   Y/U/V assembly incl. bilinear "fake" chroma + 1->16->1 mini MLPs
// Block: 256 threads (8 waves). Tile: 32x32 pixels (+halo staged in LDS).
// Each wave: 4 rows x 32 px; per row, two WMMAs (M=16 hidden, K=9 pad 32).
// ---------------------------------------------------------------------------
__global__ void __launch_bounds__(256)
fused_fullres_kernel(const float* __restrict__ full,   // (N,1,1024,1024)
                     const float* __restrict__ low,    // (N,3,256,256)
                     const float* __restrict__ cgrid,  // (N,8,16,16,6)
                     const float* __restrict__ wgd1, const float* __restrict__ bgd1,
                     const float* __restrict__ wgd2, const float* __restrict__ bgd2,
                     const float* __restrict__ wau1, const float* __restrict__ bau1,
                     const float* __restrict__ wau2, const float* __restrict__ bau2,
                     const float* __restrict__ wav1, const float* __restrict__ bav1,
                     const float* __restrict__ wav2, const float* __restrict__ bav2,
                     float* __restrict__ out)          // (N,3,1024,1024)
{
    __shared__ __align__(16) float cg[CG_FLOATS];     // 48 KB
    __shared__ float tile[TILE_ROWS * TILE_W];        // 34x34 halo, stride 36

    const int n     = blockIdx.z;
    const int tileX = blockIdx.x * 32;
    const int tileY = blockIdx.y * 32;
    const int tid   = threadIdx.x;
    const int lane  = tid & 31;
    const int wave  = tid >> 5;
    const int half  = lane >> 4;                      // 0: lanes 0-15, 1: 16-31
    const int lcol  = lane & 15;

    const float* fimg  = full + (long)n * FULL_H * FULL_W;
    const float* luimg = low + (long)(n * 3 + 1) * 256 * 256;
    const float* lvimg = low + (long)(n * 3 + 2) * 256 * 256;

    // ---- combined-grid -> LDS (async DMA path when toolchain supports it) ---
#ifdef USE_ASYNC_LDS
    {
        const float* src = cgrid + (long)n * CG_FLOATS;
        for (int i = tid * 4; i < CG_FLOATS; i += 256 * 4) {
            v4i __attribute__((address_space(1)))* g =
                (v4i __attribute__((address_space(1)))*)(uintptr_t)(src + i);
            v4i __attribute__((address_space(3)))* l =
                (v4i __attribute__((address_space(3)))*)(uintptr_t)(cg + i);
            __builtin_amdgcn_global_load_async_to_lds_b128(g, l, 0, 0);
        }
        __builtin_amdgcn_s_wait_asynccnt(0);
    }
#else
    for (int i = tid; i < CG_FLOATS; i += 256)
        cg[i] = cgrid[(long)n * CG_FLOATS + i];
#endif

    // ---- full-res tile + halo -> LDS (zero-padded at image borders) --------
    for (int i = tid; i < TILE_ROWS * 34; i += 256) {
        const int r = i / 34, c = i % 34;
        const int gy = tileY - 1 + r;
        const int gxp = tileX - 1 + c;
        float v = 0.0f;
        if (gy >= 0 && gy < FULL_H && gxp >= 0 && gxp < FULL_W)
            v = fimg[gy * FULL_W + gxp];
        tile[r * TILE_W + c] = v;
    }
    __syncthreads();

    // --- A matrix (weights, 16x9 zero-padded to 16x32), loop invariant ---
    // k order used here (same as B): K = e + 16*half; half 0 elements e=0..8 -> K=e.
    v16h A;
    for (int e = 0; e < 16; ++e) A[e] = (_Float16)0.0f;
    if (half == 0) {
        for (int e = 0; e < 9; ++e) A[e] = (_Float16)wgd1[lcol * 9 + e];
    }
    // C accumulator seeded with conv1 bias (C layout: VGPR r -> M = r + 8*half)
    v8f Cinit;
    for (int r = 0; r < 8; ++r) Cinit[r] = bgd1[r + 8 * half];
    float w2[8];
    for (int r = 0; r < 8; ++r) w2[r] = wgd2[r + 8 * half];
    const float bias2 = bgd2[0];

    for (int rr = 0; rr < 4; ++rr) {
        const int y = tileY + wave * 4 + rr;
        const int rowbase = wave * 4 + rr;            // tile row of patch ky=0

        // --- B matrices (branchless, from LDS) for pixel groups 0 and 1 ---
        // B: column N = lane&15; k order used here (same as A): K = e + 16*half.
        // Real K (0..8) lives entirely in lanes 0-15; lanes 16-31 stay zero pad.
        v16h B0, B1;
        for (int e = 0; e < 16; ++e) { B0[e] = (_Float16)0.0f; B1[e] = (_Float16)0.0f; }
        if (half == 0) {
            for (int e = 0; e < 9; ++e) {
                const int ky = e / 3, kx = e % 3;
                const float* rp = &tile[(rowbase + ky) * TILE_W + kx];
                B0[e] = (_Float16)rp[lcol];
                B1[e] = (_Float16)rp[16 + lcol];
            }
        }

        v8f acc0 = __builtin_amdgcn_wmma_f32_16x16x32_f16(
            false, A, false, B0, (short)0, Cinit, false, false);
        v8f acc1 = __builtin_amdgcn_wmma_f32_16x16x32_f16(
            false, A, false, B1, (short)0, Cinit, false, false);

        // 1x1 conv 16->1: per-lane partial dot over this half's 8 M-rows,
        // then cross-half reduction (wave32 xor-16 shuffle).
        float p0 = 0.0f, p1 = 0.0f;
        for (int r = 0; r < 8; ++r) {
            p0 += w2[r] * fmaxf(acc0[r], 0.0f);
            p1 += w2[r] * fmaxf(acc1[r], 0.0f);
        }
        const float s0 = p0 + __shfl_xor(p0, 16, 32);
        const float s1 = p1 + __shfl_xor(p1, 16, 32);
        // lane l owns pixel tileX + l: lanes<16 from group0, lanes>=16 group1
        const float g = tanhf(((half == 0) ? s0 : s1) + bias2);

        const int x = tileX + lane;

        // ---------------- trilinear slice of combined grid ----------------
        const float gx = 16.0f * (float)x / 1023.0f - 0.5f;
        const float gy = 16.0f * (float)y / 1023.0f - 0.5f;
        const float gz = 4.0f * g + 3.5f;
        const int ix0 = (int)floorf(gx);
        const int iy0 = (int)floorf(gy);
        const int iz0 = (int)floorf(gz);
        const float fx = gx - (float)ix0;
        const float fy = gy - (float)iy0;
        const float fz = gz - (float)iz0;

        float2 a01 = make_float2(0.f, 0.f);
        float2 a23 = make_float2(0.f, 0.f);
        float2 a45 = make_float2(0.f, 0.f);
        for (int dz = 0; dz < 2; ++dz)
        for (int dy = 0; dy < 2; ++dy)
        for (int dx = 0; dx < 2; ++dx) {
            const int zi = iz0 + dz, yi = iy0 + dy, xi = ix0 + dx;
            const float wz = dz ? fz : 1.0f - fz;
            const float wy = dy ? fy : 1.0f - fy;
            const float wx = dx ? fx : 1.0f - fx;
            const bool ok = (zi >= 0) & (zi < GD) & (yi >= 0) & (yi < GH) &
                            (xi >= 0) & (xi < GW);
            const float wgt = ok ? wz * wy * wx : 0.0f;
            const int zc = min(max(zi, 0), GD - 1);
            const int yc = min(max(yi, 0), GH - 1);
            const int xc = min(max(xi, 0), GW - 1);
            const float2* c2 =
                reinterpret_cast<const float2*>(&cg[((zc * GH + yc) * GW + xc) * 6]);
            const float2 v0 = c2[0], v1 = c2[1], v2 = c2[2];
            a01.x += wgt * v0.x; a01.y += wgt * v0.y;
            a23.x += wgt * v1.x; a23.y += wgt * v1.y;
            a45.x += wgt * v2.x; a45.y += wgt * v2.y;
        }

        const float pix = tile[(rowbase + 1) * TILE_W + lane + 1];
        const float Y  = pix * a01.y + a01.x;
        const float U0 = pix * a23.y + a23.x;
        const float V0 = pix * a45.y + a45.x;

        // ---- fake chroma: bilinear upsample of low_res ch1/ch2 (x4) ----
        const float sx = ((float)x + 0.5f) * 0.25f - 0.5f;
        const float sy = ((float)y + 0.5f) * 0.25f - 0.5f;
        int lx0 = (int)floorf(sx), ly0 = (int)floorf(sy);
        const float ax = sx - (float)lx0;
        const float ay = sy - (float)ly0;
        int lx1 = min(max(lx0 + 1, 0), 255), ly1 = min(max(ly0 + 1, 0), 255);
        lx0 = min(max(lx0, 0), 255);
        ly0 = min(max(ly0, 0), 255);
        const float fU =
            (1.0f - ay) * ((1.0f - ax) * luimg[ly0 * 256 + lx0] + ax * luimg[ly0 * 256 + lx1]) +
            ay          * ((1.0f - ax) * luimg[ly1 * 256 + lx0] + ax * luimg[ly1 * 256 + lx1]);
        const float fV =
            (1.0f - ay) * ((1.0f - ax) * lvimg[ly0 * 256 + lx0] + ax * lvimg[ly0 * 256 + lx1]) +
            ay          * ((1.0f - ax) * lvimg[ly1 * 256 + lx0] + ax * lvimg[ly1 * 256 + lx1]);

        // ---- 1->16->1 chroma MLPs ----
        float au = 0.0f, av = 0.0f;
        for (int k = 0; k < 16; ++k) {
            au += wau2[k] * fmaxf(wau1[k] * U0 + bau1[k], 0.0f);
            av += wav2[k] * fmaxf(wav1[k] * V0 + bav1[k], 0.0f);
        }
        const float U = tanhf(au + bau2[0]) + fU;
        const float V = tanhf(av + bav2[0]) + fV;

        out[((long)(n * 3 + 0) * FULL_H + y) * FULL_W + x] = Y;
        out[((long)(n * 3 + 1) * FULL_H + y) * FULL_W + x] = U;
        out[((long)(n * 3 + 2) * FULL_H + y) * FULL_W + x] = V;
    }
}

// ---------------------------------------------------------------------------
// Host launch
// ---------------------------------------------------------------------------
extern "C" void kernel_launch(void* const* d_in, const int* in_sizes, int n_in,
                              void* d_out, int out_size, void* d_ws, size_t ws_size,
                              hipStream_t stream) {
    const float* low   = (const float*)d_in[0];
    const float* full  = (const float*)d_in[1];
    const float* w0 = (const float*)d_in[2];  const float* b0 = (const float*)d_in[3];
    const float* w1 = (const float*)d_in[4];  const float* b1 = (const float*)d_in[5];
    const float* w2 = (const float*)d_in[6];  const float* b2 = (const float*)d_in[7];
    const float* w3 = (const float*)d_in[8];  const float* b3 = (const float*)d_in[9];
    const float* wl0 = (const float*)d_in[10]; const float* bl0 = (const float*)d_in[11];
    const float* wl1 = (const float*)d_in[12];
    const float* wg1 = (const float*)d_in[13]; const float* bg1 = (const float*)d_in[14];
    const float* wg2 = (const float*)d_in[15]; const float* bg2 = (const float*)d_in[16];
    const float* wf1 = (const float*)d_in[17]; const float* bf1 = (const float*)d_in[18];
    const float* wf2 = (const float*)d_in[19]; const float* bf2 = (const float*)d_in[20];
    const float* wf3 = (const float*)d_in[21]; const float* bf3 = (const float*)d_in[22];
    const float* wlin = (const float*)d_in[23]; const float* blin = (const float*)d_in[24];
    const float* wgd1 = (const float*)d_in[25]; const float* bgd1 = (const float*)d_in[26];
    const float* wgd2 = (const float*)d_in[27]; const float* bgd2 = (const float*)d_in[28];
    const float* wau1 = (const float*)d_in[29]; const float* bau1 = (const float*)d_in[30];
    const float* wau2 = (const float*)d_in[31]; const float* bau2 = (const float*)d_in[32];
    const float* wav1 = (const float*)d_in[33]; const float* bav1 = (const float*)d_in[34];
    const float* wav2 = (const float*)d_in[35]; const float* bav2 = (const float*)d_in[36];

    float* ws = (float*)d_ws;
    // workspace layout (floats)
    float* x0    = ws;                       // 4*8*128*128   = 524288
    float* x1    = x0    + 524288;           // 4*16*64*64    = 262144
    float* x2    = x1    + 262144;           // 4*32*32*32    = 131072
    float* splat = x2    + 131072;           // 4*64*16*16    = 65536
    float* lh    = splat + 65536;            // 4*64*16*16    = 65536
    float* localb= lh    + 65536;            // 4*64*16*16    = 65536
    float* g1b   = localb+ 65536;            // 4*128*8*8     = 32768
    float* g2b   = g1b   + 32768;            // 4*256*4*4     = 16384
    float* cgrid = g2b   + 16384;            // 4*8*16*16*6   = 49152

    auto blocks = [](int total) { return (total + 255) / 256; };

    // Splat stack (stride-2 convs)
    conv3x3_kernel<<<blocks(4*8*128*128), 256, 0, stream>>>(
        low, w0, b0, x0, 4, 3, 256, 256, 8, 128, 128, 2, 1);
    conv3x3_kernel<<<blocks(4*16*64*64), 256, 0, stream>>>(
        x0, w1, b1, x1, 4, 8, 128, 128, 16, 64, 64, 2, 1);
    conv3x3_kernel<<<blocks(4*32*32*32), 256, 0, stream>>>(
        x1, w2, b2, x2, 4, 16, 64, 64, 32, 32, 32, 2, 1);
    conv3x3_kernel<<<blocks(4*64*16*16), 256, 0, stream>>>(
        x2, w3, b3, splat, 4, 32, 32, 32, 64, 16, 16, 2, 1);
    // Local branch
    conv3x3_kernel<<<blocks(4*64*16*16), 256, 0, stream>>>(
        splat, wl0, bl0, lh, 4, 64, 16, 16, 64, 16, 16, 1, 1);
    conv3x3_kernel<<<blocks(4*64*16*16), 256, 0, stream>>>(
        lh, wl1, nullptr, localb, 4, 64, 16, 16, 64, 16, 16, 1, 0);
    // Global branch convs
    conv3x3_kernel<<<blocks(4*128*8*8), 256, 0, stream>>>(
        splat, wg1, bg1, g1b, 4, 64, 16, 16, 128, 8, 8, 2, 1);
    conv3x3_kernel<<<blocks(4*256*4*4), 256, 0, stream>>>(
        g1b, wg2, bg2, g2b, 4, 128, 8, 8, 256, 4, 4, 2, 1);
    // Means + FC chain + fuse + grid combine
    global_tail_kernel<<<4, 256, 0, stream>>>(
        splat, g1b, g2b, localb,
        wf1, bf1, wf2, bf2, wf3, bf3, wlin, blin, cgrid);
    // Fused full-res: guide (WMMA) + slice + output assembly
    fused_fullres_kernel<<<dim3(32, 32, 4), 256, 0, stream>>>(
        full, low, cgrid,
        wgd1, bgd1, wgd2, bgd2,
        wau1, bau1, wau2, bau2,
        wav1, bav1, wav2, bav2,
        (float*)d_out);
}
